// FBSNN_16836271800856
// MI455X (gfx1250) — matrix operand order repaired and checked by reading, not compiled, not run on hardware
//
#include <hip/hip_runtime.h>

// ---------------------------------------------------------------------------
// FBSNN loss on MI455X (gfx1250): f16 WMMA (16x16x32, f32 accum) MLP fwd+VJP
// fused with the sequential Euler-Maruyama stepper. Compute-bound (~108 GFLOP,
// ~9 MB HBM traffic) -> all heavy math on v_wmma_f32_16x16x32_f16.
// Round-2 fixes: rolled K-loop (kills scratch spills around v_wmma),
// rcp-based tanh (kills v_div_* sequences in the epilogue).
// ---------------------------------------------------------------------------

typedef __attribute__((ext_vector_type(16))) _Float16 v16h;
typedef __attribute__((ext_vector_type(8)))  _Float16 v8h;
typedef __attribute__((ext_vector_type(8)))  float    v8f;

#define HIDN   256
#define DIMN   16
#define NSTEP  64
#define BATCH  2048
#define MTILE  32
#define NWG    (BATCH / MTILE)
#define XS     40      // LDS row stride (halves) for the 32-wide input tile
#define HS     264     // LDS row stride (halves) for 256-wide activations (bank-conflict-free)
#define NTIL   16      // number of 16-col tiles in a 256-wide layer
#define SIGMA  0.3f

// -------- pre-swizzled f16 weight-fragment workspace layout (in halves) -----
#define OFF_W1F   0
#define SZ_W1F    (32 * HIDN)        // W1 zero-padded to K=32
#define OFF_W2F   (OFF_W1F + SZ_W1F)
#define SZ_WF     (HIDN * HIDN)
#define OFF_W2TF  (OFF_W2F + 3 * SZ_WF)
#define OFF_W1TY  (OFF_W2TF + 3 * SZ_WF)
#define SZ_W1TY   (HIDN * 16)        // W1[1:17,:]^T  (256 x 16)
#define TOTAL_HALVES (OFF_W1TY + SZ_W1TY)

// ---------------------------------------------------------------------------
// Prep: pack f32 weights into a contiguous-k 16-bit B-matrix (32x16 tile) fragment
// layout.  Within a tile: lane l holds column n = l&15; K-local = (l&16?16:0)+s
// for half-slot s = 0..15; the 16 halves of a lane are stored contiguously so
// the GEMM loads one 32-byte vector per lane per tile.
// ---------------------------------------------------------------------------
__global__ __launch_bounds__(256) void prep_weights(
    const float* __restrict__ W1, const float* __restrict__ W2,
    const float* __restrict__ W3, const float* __restrict__ W4,
    _Float16* __restrict__ wf)
{
  int idx = blockIdx.x * blockDim.x + threadIdx.x;
  if (idx >= TOTAL_HALVES) return;
  float val;
  if (idx < OFF_W2F) {                      // W1 padded to 32 x 256
    int u = idx;
    int tile = u >> 9, l = (u >> 4) & 31, s = u & 15;
    int kk = tile >> 4, nb = tile & 15;
    int k = kk * 32 + ((l & 16) ? 16 : 0) + s;
    int n = nb * 16 + (l & 15);
    val = (k < DIMN + 1) ? W1[k * HIDN + n] : 0.0f;
  } else if (idx < OFF_W2TF) {              // W2, W3, W4 (forward)
    int u = idx - OFF_W2F;
    int m = u / SZ_WF; u %= SZ_WF;
    const float* W = (m == 0) ? W2 : ((m == 1) ? W3 : W4);
    int tile = u >> 9, l = (u >> 4) & 31, s = u & 15;
    int kk = tile >> 4, nb = tile & 15;
    int k = kk * 32 + ((l & 16) ? 16 : 0) + s;
    int n = nb * 16 + (l & 15);
    val = W[k * HIDN + n];
  } else if (idx < OFF_W1TY) {              // W2^T, W3^T, W4^T (backward)
    int u = idx - OFF_W2TF;
    int m = u / SZ_WF; u %= SZ_WF;
    const float* W = (m == 0) ? W2 : ((m == 1) ? W3 : W4);
    int tile = u >> 9, l = (u >> 4) & 31, s = u & 15;
    int kk = tile >> 4, nb = tile & 15;
    int k = kk * 32 + ((l & 16) ? 16 : 0) + s;
    int n = nb * 16 + (l & 15);
    val = W[n * HIDN + k];
  } else {                                  // W1[1:17,:]^T : 256 x 16 (for dY)
    int u = idx - OFF_W1TY;
    int kk = u >> 9, l = (u >> 4) & 31, s = u & 15;
    int k = kk * 32 + ((l & 16) ? 16 : 0) + s;
    int n = l & 15;
    val = W1[(n + 1) * HIDN + k];
  }
  wf[idx] = (_Float16)val;
}

// tanh(|x|) = (1-e)/(1+e), e = exp(-2|x|); use raw v_rcp_f32 + v_exp_f32
// (TRANS pipe, co-executes with WMMA) instead of IEEE division.
__device__ __forceinline__ float fast_tanh(float x) {
  float ax = fabsf(x);
  float e  = __expf(-2.0f * ax);
  float t  = (1.0f - e) * __builtin_amdgcn_rcpf(1.0f + e);
  return copysignf(t, x);
}

// ---------------------------------------------------------------------------
// One 32x256 (K = 32*KK) GEMM layer on 8 waves: wave owns rows rb*16..+15 and
// four 16-col tiles.  A fragments from LDS (layout: lanes<16 hold K 0-15;
// lanes>=16 hold K 16-31 -> two ds 128-bit loads), B fragments
// pre-swizzled contiguous in global.  FWD: +bias, tanh.  !FWD: * (1 - h^2).
// K-loop deliberately rolled to keep one A + four B fragments in flight.
// ---------------------------------------------------------------------------
template <int KK, bool FWD>
__device__ __forceinline__ void gemm_layer(
    const _Float16* __restrict__ A, int As,
    const _Float16* __restrict__ Bf,
    const float*    __restrict__ bias,
    const _Float16* __restrict__ Hprev,
    _Float16*       __restrict__ Out,
    int wave, int lane)
{
  const int rb   = wave & 1;
  const int cg   = wave >> 1;            // 0..3, each covers 4 col tiles
  const int mrow = rb * 16 + (lane & 15);
  const int lhi  = lane >> 4;
  v8f acc[4];
#pragma unroll
  for (int t = 0; t < 4; ++t)
#pragma unroll
    for (int r = 0; r < 8; ++r) acc[t][r] = 0.0f;

#pragma unroll 1
  for (int kk = 0; kk < KK; ++kk) {
    const _Float16* ap = A + mrow * As + kk * 32 + lhi * 16;
    v8h a0 = *(const v8h*)(ap);
    v8h a1 = *(const v8h*)(ap + 8);
    v16h a = __builtin_shufflevector(a0, a1, 0, 1, 2, 3, 4, 5, 6, 7,
                                             8, 9, 10, 11, 12, 13, 14, 15);
    const v16h* bp =
        (const v16h*)(Bf + (size_t)((kk * NTIL + cg * 4) * 32 + lane) * 16);
#pragma unroll
    for (int t = 0; t < 4; ++t) {
      acc[t] = __builtin_amdgcn_wmma_f32_16x16x32_f16(
          false, a, false, bp[t * 32], (short)0, acc[t], false, false);
    }
  }
#pragma unroll
  for (int t = 0; t < 4; ++t) {
    int n = (cg * 4 + t) * 16 + (lane & 15);
    float bv = 0.0f;
    if constexpr (FWD) bv = bias[n];
#pragma unroll
    for (int r = 0; r < 8; ++r) {
      int m = rb * 16 + r + lhi * 8;     // D layout: vgpr r, lane-half selects M+8
      float v = acc[t][r] + bv;
      if constexpr (FWD) {
        v = fast_tanh(v);
      } else {
        float h = (float)Hprev[m * HS + n];
        v = v * (1.0f - h * h);
      }
      Out[m * HS + n] = (_Float16)v;
    }
  }
}

// dY = g1 (32x256) @ W1y^T-frags (256x16) -> 32x16 f32 in LDS; waves 0-1 only.
__device__ __forceinline__ void gemm_dY(
    const _Float16* __restrict__ G, const _Float16* __restrict__ Wty,
    float* __restrict__ dY, int wave, int lane)
{
  if (wave >= 2) return;
  const int rb   = wave;
  const int mrow = rb * 16 + (lane & 15);
  const int lhi  = lane >> 4;
  v8f acc;
#pragma unroll
  for (int r = 0; r < 8; ++r) acc[r] = 0.0f;
#pragma unroll 1
  for (int kk = 0; kk < 8; ++kk) {
    const _Float16* ap = G + mrow * HS + kk * 32 + lhi * 16;
    v8h a0 = *(const v8h*)(ap);
    v8h a1 = *(const v8h*)(ap + 8);
    v16h a = __builtin_shufflevector(a0, a1, 0, 1, 2, 3, 4, 5, 6, 7,
                                             8, 9, 10, 11, 12, 13, 14, 15);
    const v16h* bp = (const v16h*)(Wty + (size_t)(kk * 32 + lane) * 16);
    acc = __builtin_amdgcn_wmma_f32_16x16x32_f16(
        false, a, false, *bp, (short)0, acc, false, false);
  }
  int n = lane & 15;
#pragma unroll
  for (int r = 0; r < 8; ++r) {
    int m = rb * 16 + r + lhi * 8;
    dY[m * DIMN + n] = acc[r];
  }
}

// ---------------------------------------------------------------------------
// Main: one workgroup per 32 batch rows; sequential loop over 65 time points.
// ---------------------------------------------------------------------------
__global__ __launch_bounds__(256) void fbsnn_main(
    const float* __restrict__ tP, const float* __restrict__ WP,
    const float* __restrict__ y0v,
    const float* __restrict__ b1, const float* __restrict__ b2,
    const float* __restrict__ b3, const float* __restrict__ b4,
    const float* __restrict__ W5, const float* __restrict__ b5,
    const _Float16* __restrict__ wf, float* __restrict__ partial)
{
  extern __shared__ __align__(16) char smem[];
  _Float16* X  = (_Float16*)smem;                 // MTILE * XS
  _Float16* H0 = X  + MTILE * XS;                 // activations (post-tanh)
  _Float16* H1 = H0 + MTILE * HS;
  _Float16* H2 = H1 + MTILE * HS;
  _Float16* H3 = H2 + MTILE * HS;
  _Float16* G0 = H3 + MTILE * HS;                 // backprop ping-pong
  _Float16* G1 = G0 + MTILE * HS;
  float* yb   = (float*)(G1 + MTILE * HS);        // MTILE x DIM state
  float* dY   = yb   + MTILE * DIMN;              // MTILE x DIM gradient
  float* Yv   = dY   + MTILE * DIMN;              // MTILE net outputs
  float* Ytil = Yv   + MTILE;                     // MTILE predicted Y~
  float* red  = Ytil + MTILE;                     // 256 reduction slots

  const int tid  = threadIdx.x;
  const int wave = tid >> 5, lane = tid & 31;
  const int row0 = blockIdx.x * MTILE;
  float loss = 0.0f;

  const _Float16* W1f  = wf + OFF_W1F;
  const _Float16* W2f  = wf + OFF_W2F;
  const _Float16* W3f  = wf + OFF_W2F + SZ_WF;
  const _Float16* W4f  = wf + OFF_W2F + 2 * SZ_WF;
  const _Float16* W2tf = wf + OFF_W2TF;
  const _Float16* W3tf = wf + OFF_W2TF + SZ_WF;
  const _Float16* W4tf = wf + OFF_W2TF + 2 * SZ_WF;
  const _Float16* W1ty = wf + OFF_W1TY;

  for (int idx = tid; idx < MTILE * DIMN; idx += 256)
    yb[idx] = y0v[idx & (DIMN - 1)];
  __syncthreads();

#pragma unroll 1
  for (int i = 0; i <= NSTEP; ++i) {
    // ---- build X = [t, y, 0-pad] in f16 --------------------------------
    for (int idx = tid; idx < MTILE * 32; idx += 256) {
      int r = idx >> 5, c = idx & 31;
      float v;
      if (c == 0)            v = tP[(size_t)(row0 + r) * (NSTEP + 1) + i];
      else if (c <= DIMN)    v = yb[r * DIMN + (c - 1)];
      else                   v = 0.0f;
      X[r * XS + c] = (_Float16)v;
    }
    __syncthreads();

    // ---- forward -------------------------------------------------------
    gemm_layer<1, true>(X,  XS, W1f, b1, nullptr, H0, wave, lane); __syncthreads();
    gemm_layer<8, true>(H0, HS, W2f, b2, nullptr, H1, wave, lane); __syncthreads();
    gemm_layer<8, true>(H1, HS, W3f, b3, nullptr, H2, wave, lane); __syncthreads();
    gemm_layer<8, true>(H2, HS, W4f, b4, nullptr, H3, wave, lane); __syncthreads();

    // ---- output layer + backprop seed g4 = w5 * (1 - h4^2) -------------
    if (tid < MTILE) {
      float acc = b5[0];
      for (int n = 0; n < HIDN; ++n) acc += (float)H3[tid * HS + n] * W5[n];
      Yv[tid] = acc;
    }
    for (int idx = tid; idx < MTILE * HIDN; idx += 256) {
      int r = idx >> 8, n = idx & 255;
      float h = (float)H3[r * HS + n];
      G0[r * HS + n] = (_Float16)(W5[n] * (1.0f - h * h));
    }
    __syncthreads();

    // ---- backward (VJP) ------------------------------------------------
    gemm_layer<8, false>(G0, HS, W4tf, nullptr, H2, G1, wave, lane); __syncthreads();
    gemm_layer<8, false>(G1, HS, W3tf, nullptr, H1, G0, wave, lane); __syncthreads();
    gemm_layer<8, false>(G0, HS, W2tf, nullptr, H0, G1, wave, lane); __syncthreads();
    gemm_dY(G1, W1ty, dY, wave, lane);
    __syncthreads();

    // ---- Euler-Maruyama step + loss ------------------------------------
    if (tid < MTILE) {
      const int b = row0 + tid;
      const float Y = Yv[tid];
      if (i > 0) { float d = Y - Ytil[tid]; loss += d * d; }
      if (i < NSTEP) {
        float t0 = tP[(size_t)b * (NSTEP + 1) + i];
        float t1 = tP[(size_t)b * (NSTEP + 1) + i + 1];
        float dt = t1 - t0;
        float th = fast_tanh(Y);
        float qq = 0.0f, yy = 0.0f, zdw = 0.0f;
#pragma unroll
        for (int d = 0; d < DIMN; ++d) {
          float y  = yb[tid * DIMN + d];
          float w0 = WP[((size_t)b * (NSTEP + 1) + i)     * DIMN + d];
          float w1 = WP[((size_t)b * (NSTEP + 1) + i + 1) * DIMN + d];
          float dW = w1 - w0;
          float q  = -y * th;
          qq  += q * q;
          yy  += y * y;
          zdw += dY[tid * DIMN + d] * dW;
          yb[tid * DIMN + d] = y + q * dt + SIGMA * dW;
        }
        Ytil[tid] = Y - 0.5f * (qq + yy) * dt + SIGMA * zdw;
      } else {
        float yy = 0.0f;
#pragma unroll
        for (int d = 0; d < DIMN; ++d) { float y = yb[tid * DIMN + d]; yy += y * y; }
        float dT = Y - yy; loss += dT * dT;
#pragma unroll
        for (int d = 0; d < DIMN; ++d) {
          float g = dY[tid * DIMN + d] - 2.0f * yb[tid * DIMN + d];
          loss += g * g;
        }
      }
    }
    __syncthreads();
  }

  // ---- deterministic block reduction -> workspace partials ----------------
  red[tid] = loss;
  __syncthreads();
  for (int s = 128; s > 0; s >>= 1) {
    if (tid < s) red[tid] += red[tid + s];
    __syncthreads();
  }
  if (tid == 0) partial[blockIdx.x] = red[0];
}

__global__ void finalize_loss(const float* __restrict__ partial, float* __restrict__ out) {
  float s = 0.0f;
  for (int i = 0; i < NWG; ++i) s += partial[i];
  out[0] = s;
}

extern "C" void kernel_launch(void* const* d_in, const int* in_sizes, int n_in,
                              void* d_out, int out_size, void* d_ws, size_t ws_size,
                              hipStream_t stream) {
  (void)in_sizes; (void)n_in; (void)out_size; (void)ws_size;
  const float* tP = (const float*)d_in[0];
  const float* WP = (const float*)d_in[1];
  const float* y0 = (const float*)d_in[2];
  const float* W1 = (const float*)d_in[3];
  const float* b1 = (const float*)d_in[4];
  const float* W2 = (const float*)d_in[5];
  const float* b2 = (const float*)d_in[6];
  const float* W3 = (const float*)d_in[7];
  const float* b3 = (const float*)d_in[8];
  const float* W4 = (const float*)d_in[9];
  const float* b4 = (const float*)d_in[10];
  const float* W5 = (const float*)d_in[11];
  const float* b5 = (const float*)d_in[12];

  _Float16* wf = (_Float16*)d_ws;
  float* partial = (float*)((char*)d_ws + (size_t)TOTAL_HALVES * sizeof(_Float16));
  float* out = (float*)d_out;

  prep_weights<<<(TOTAL_HALVES + 255) / 256, 256, 0, stream>>>(W1, W2, W3, W4, wf);

  const size_t smem =
      (size_t)(MTILE * XS + 6 * MTILE * HS) * sizeof(_Float16) +
      (size_t)(2 * MTILE * DIMN + 2 * MTILE + 256) * sizeof(float);
  (void)hipFuncSetAttribute((const void*)fbsnn_main,
                            hipFuncAttributeMaxDynamicSharedMemorySize,
                            (int)smem);
  fbsnn_main<<<NWG, 256, smem, stream>>>(tP, WP, y0, b1, b2, b3, b4, W5, b5,
                                         wf, partial);
  finalize_loss<<<1, 1, 0, stream>>>(partial, out);
}
